// NodeGat_42107859370482
// MI455X (gfx1250) — matrix operand order repaired and checked by reading, not compiled, not run on hardware
//
#include <hip/hip_runtime.h>
#include <hip/hip_bf16.h>

typedef __attribute__((ext_vector_type(16))) _Float16 v16h;
typedef __attribute__((ext_vector_type(8)))  _Float16 v8h;
typedef __attribute__((ext_vector_type(8)))  float    v8f;
typedef __attribute__((ext_vector_type(4)))  int      i4;
typedef __attribute__((address_space(1))) i4* gi4p;   // global int4*
typedef __attribute__((address_space(3))) i4* li4p;   // LDS int4*

#if defined(__has_builtin)
#  if __has_builtin(__builtin_amdgcn_global_load_async_to_lds_b128)
#    define HAVE_ASYNC_LDS 1
#  endif
#endif
#ifndef HAVE_ASYNC_LDS
#  define HAVE_ASYNC_LDS 0
#endif

// address-space casts via inttoptr (avoids addrspacecast diagnostics)
__device__ __forceinline__ gi4p as_global_i4(const void* p) {
  return (gi4p)(unsigned long long)p;
}
__device__ __forceinline__ li4p as_lds_i4(const void* p) {
  return (li4p)(unsigned)(unsigned long long)p;   // LDS offset = low 32 bits of flat addr
}

// ---------- helpers ----------
__device__ __forceinline__ unsigned encf(float f) {
  unsigned u = __float_as_uint(f);
  return (u & 0x80000000u) ? ~u : (u | 0x80000000u);
}
__device__ __forceinline__ float decf(unsigned e) {
  return (e & 0x80000000u) ? __uint_as_float(e & 0x7FFFFFFFu)
                           : __uint_as_float(~e);
}

// ---------- conversion kernels ----------
__global__ void k_f32_to_f16(const float* __restrict__ in, _Float16* __restrict__ out, int n) {
  int i = blockIdx.x * blockDim.x + threadIdx.x;
  if (i < n) out[i] = (_Float16)in[i];
}

// W [128, Nc] f32 (row-major) -> WT [Nc, 128] f16 (transposed)
__global__ void k_w_to_f16_T(const float* __restrict__ in, _Float16* __restrict__ out, int Nc) {
  int i = blockIdx.x * blockDim.x + threadIdx.x;   // over 128*Nc
  if (i >= 128 * Nc) return;
  int k = i / Nc, j = i - k * Nc;
  out[j * 128 + k] = (_Float16)in[i];
}

// Wc [128,40] f32 -> WCT [48,128] f16 transposed, rows 40..47 zero
__global__ void k_pad_wc_T(const float* __restrict__ wc, _Float16* __restrict__ out) {
  int i = blockIdx.x * blockDim.x + threadIdx.x;   // over 48*128
  if (i >= 48 * 128) return;
  int j = i / 128, k = i - j * 128;                // j = out col (class), k = K
  out[i] = (j < 40) ? (_Float16)wc[k * 40 + j] : (_Float16)0.0f;
}

// ---------- WMMA GEMM: C[M, CT*16] = A[M,128](f16) * B (given as BT[CT*16,128] f16) ----
// One wave per 16-row stripe; accumulates all CT column tiles (A fragment reused CT x).
// BT is staged in LDS via async global->LDS copy when available.
template <int CT>
__global__ void gat_wmma_gemm(const _Float16* __restrict__ A,
                              const _Float16* __restrict__ BT,
                              float* __restrict__ C,
                              int rowTiles) {
  constexpr int K   = 128;
  constexpr int LDB = K + 8;                       // pad 8 halves -> bank spread
  constexpr int NC  = CT * 16;
  __shared__ _Float16 Bs[NC * LDB];

  // cooperative copy of BT (NC x 128 f16) into LDS, 16B chunks
  for (int c = threadIdx.x; c < NC * (K / 8); c += blockDim.x) {
    int row = c >> 4, seg = c & 15;
#if HAVE_ASYNC_LDS
    __builtin_amdgcn_global_load_async_to_lds_b128(
        as_global_i4(BT + row * K + seg * 8),
        as_lds_i4(&Bs[row * LDB + seg * 8]),
        0, 0);
#else
    *(v8h*)&Bs[row * LDB + seg * 8] = *(const v8h*)(BT + row * K + seg * 8);
#endif
  }
#if HAVE_ASYNC_LDS
#  if defined(__has_builtin) && __has_builtin(__builtin_amdgcn_s_wait_asynccnt)
  __builtin_amdgcn_s_wait_asynccnt(0);
#  else
  asm volatile("s_wait_asynccnt 0x0" ::: "memory");
#  endif
#endif
  __syncthreads();

  int gwave = (blockIdx.x * blockDim.x + threadIdx.x) >> 5;  // row-tile id
  if (gwave >= rowTiles) return;                             // wave-uniform
  int lane = threadIdx.x & 31;
  int half = lane >> 4;                                      // lane group
  int l    = lane & 15;

  const _Float16* arow = A + (size_t)(gwave * 16 + l) * K;

  v8f acc[CT];
#pragma unroll
  for (int t = 0; t < CT; ++t) acc[t] = (v8f){};

#pragma unroll
  for (int k0 = 0; k0 < K; k0 += 32) {
    // A 16x32 f16: lanes0-15 K {k0..k0+15}; lanes16-31 K {k0+16..k0+31}
    v8h alo = *(const v8h*)(arow + k0 + half * 16);
    v8h ahi = *(const v8h*)(arow + k0 + half * 16 + 8);
    v16h a;
#pragma unroll
    for (int i = 0; i < 8; ++i) { a[i] = alo[i]; a[8 + i] = ahi[i]; }

    int kb = k0 + half * 16;                                 // B K-range per lane group
#pragma unroll
    for (int t = 0; t < CT; ++t) {
      const _Float16* bp = &Bs[(t * 16 + l) * LDB + kb];     // contiguous 16 K values
      v8h b0 = *(const v8h*)bp;
      v8h b1 = *(const v8h*)(bp + 8);
      v16h b;
#pragma unroll
      for (int i = 0; i < 8; ++i) { b[i] = b0[i]; b[8 + i] = b1[i]; }
      acc[t] = __builtin_amdgcn_wmma_f32_16x16x32_f16(false, a, false, b,
                                                      (short)0, acc[t], false, false);
    }
  }

  // C/D 16x16 f32: VGPR r -> M = r (lanes 0-15) / 8+r (lanes 16-31); N = lane&15
  int mrow = gwave * 16 + half * 8;
#pragma unroll
  for (int t = 0; t < CT; ++t) {
    int ncol = t * 16 + l;
#pragma unroll
    for (int r = 0; r < 8; ++r)
      C[(size_t)(mrow + r) * NC + ncol] = acc[t][r];
  }
}

// ---------- attention dot products: a_src[n,h], a_dst[n,h] ----------
__global__ void k_att(const float* __restrict__ xh,
                      const float* __restrict__ att_s, const float* __restrict__ att_d,
                      float* __restrict__ asrc, float* __restrict__ adst, int N) {
  int t = blockIdx.x * blockDim.x + threadIdx.x;   // t = n*4 + h
  if (t >= N * 4) return;
  int n = t >> 2, h = t & 3;
  const float* p = xh + (size_t)n * 128 + h * 32;
  float s = 0.f, d = 0.f;
#pragma unroll
  for (int c = 0; c < 32; ++c) {
    s += p[c] * att_s[h * 32 + c];
    d += p[c] * att_d[h * 32 + c];
  }
  asrc[t] = s;
  adst[t] = d;
}

// ---------- edge phase ----------
__global__ void k_edge_logit(const long long* __restrict__ ei, int Er, int Et,
                             const float* __restrict__ asrc, const float* __restrict__ adst,
                             float* __restrict__ logits, unsigned* __restrict__ menc) {
  int idx = blockIdx.x * blockDim.x + threadIdx.x; // over Et*4
  if (idx >= Et * 4) return;
  int e = idx >> 2, h = idx & 3;
  long long s, d;
  if (e < Er) { s = ei[e]; d = ei[Er + e]; } else { s = e - Er; d = s; }
  float v = asrc[s * 4 + h] + adst[d * 4 + h];
  v = (v > 0.f) ? v : 0.2f * v;                    // leaky_relu(0.2)
  logits[idx] = v;
  atomicMax(menc + d * 4 + h, encf(v));
}

__global__ void k_edge_exp(const long long* __restrict__ ei, int Er, int Et,
                           float* __restrict__ logits, const unsigned* __restrict__ menc,
                           float* __restrict__ den) {
  int idx = blockIdx.x * blockDim.x + threadIdx.x;
  if (idx >= Et * 4) return;
  int e = idx >> 2, h = idx & 3;
  long long d;
  if (e < Er) { d = ei[Er + e]; } else { d = e - Er; }
  float m = decf(menc[d * 4 + h]);
  float ex = __expf(logits[idx] - m);
  logits[idx] = ex;
  atomicAdd(den + d * 4 + h, ex);
}

// one block per edge; 128 threads = (head h = t>>5, channel c = t&31)
__global__ void k_edge_scatter(const long long* __restrict__ ei, int Er,
                               const float* __restrict__ xh, const float* __restrict__ ex,
                               const float* __restrict__ den, float* __restrict__ acc) {
  int e = blockIdx.x;
  int t = threadIdx.x;
  int h = t >> 5;
  long long s, d;
  if (e < Er) { s = ei[e]; d = ei[Er + e]; } else { s = e - Er; d = s; }
  float alpha = ex[(size_t)e * 4 + h] / (den[d * 4 + h] + 1e-16f);
  atomicAdd(&acc[(size_t)d * 128 + t], xh[(size_t)s * 128 + t] * alpha);
}

// ---------- bias + relu, emit f16 input for next GEMM ----------
__global__ void k_bias_relu_f16(const float* __restrict__ acc, const float* __restrict__ bias,
                                _Float16* __restrict__ out16, int total) {
  int i = blockIdx.x * blockDim.x + threadIdx.x;
  if (i >= total) return;
  float v = acc[i] + bias[i & 127];
  v = (v > 0.f) ? v : 0.f;
  out16[i] = (_Float16)v;
}

// ---------- classifier log-softmax (reads padded [N,48], writes [N,40]) ----------
__global__ void k_logsoftmax(const float* __restrict__ lpad, const float* __restrict__ bc,
                             float* __restrict__ out, int N) {
  int n = blockIdx.x * blockDim.x + threadIdx.x;
  if (n >= N) return;
  float v[40];
  float mx = -3.0e38f;
#pragma unroll
  for (int j = 0; j < 40; ++j) {
    v[j] = lpad[(size_t)n * 48 + j] + bc[j];
    mx = fmaxf(mx, v[j]);
  }
  float s = 0.f;
#pragma unroll
  for (int j = 0; j < 40; ++j) s += __expf(v[j] - mx);
  float ls = __logf(s) + mx;
#pragma unroll
  for (int j = 0; j < 40; ++j) out[(size_t)n * 40 + j] = v[j] - ls;
}

extern "C" void kernel_launch(void* const* d_in, const int* in_sizes, int n_in,
                              void* d_out, int out_size, void* d_ws, size_t ws_size,
                              hipStream_t stream) {
  const float*     x   = (const float*)d_in[0];
  const long long* ei  = (const long long*)d_in[1];
  const float*     W1  = (const float*)d_in[2];
  const float*     as1 = (const float*)d_in[3];
  const float*     ad1 = (const float*)d_in[4];
  const float*     b1  = (const float*)d_in[5];
  const float*     W2  = (const float*)d_in[6];
  const float*     as2 = (const float*)d_in[7];
  const float*     ad2 = (const float*)d_in[8];
  const float*     b2  = (const float*)d_in[9];
  const float*     Wc  = (const float*)d_in[10];
  const float*     bc  = (const float*)d_in[11];
  float* out = (float*)d_out;

  const int N  = in_sizes[0] / 128;   // 50000 (multiple of 16)
  const int Er = in_sizes[1] / 2;     // 800000
  const int Et = Er + N;              // + self loops

  // ---- workspace layout (256B aligned slabs) ----
  char* w = (char*)d_ws;
  auto alloc = [&](size_t bytes) {
    char* p = w;
    w += (bytes + 255) & ~(size_t)255;
    return p;
  };
  _Float16* X16  = (_Float16*)alloc((size_t)N * 128 * 2);  // GEMM A operand (f16)
  _Float16* WT16 = (_Float16*)alloc((size_t)128 * 128 * 2);// transposed weight (f16)
  _Float16* WCT  = (_Float16*)alloc((size_t)48 * 128 * 2); // transposed padded classifier W
  float*    XH   = (float*)alloc((size_t)N * 128 * 4);     // xh = X@W (f32)
  float*    ASRC = (float*)alloc((size_t)N * 4 * 4);
  float*    ADST = (float*)alloc((size_t)N * 4 * 4);
  unsigned* MENC = (unsigned*)alloc((size_t)N * 4 * 4);
  float*    DEN  = (float*)alloc((size_t)N * 4 * 4);
  float*    EXL  = (float*)alloc((size_t)Et * 4 * 4);      // per-edge logits -> exp
  float*    ACC  = (float*)alloc((size_t)N * 128 * 4);     // segment-sum accumulator
  float*    LPAD = (float*)alloc((size_t)N * 48 * 4);      // padded classifier logits

  const int T = 256;
  const int rowTiles = N / 16;                             // 3125
  const int gemmBlk = (rowTiles + 7) / 8;                  // 8 waves/block
  const int nhBlk = (N * 4 + T - 1) / T;
  const int ehBlk = (Et * 4 + T - 1) / T;
  const int nfBlk = (N * 128 + T - 1) / T;

  auto gat_layer = [&](const _Float16* A16, const float* Wf, const float* as,
                       const float* ad, const float* bias) {
    // xh = A16 @ W (WMMA, B transposed + LDS-staged)
    k_w_to_f16_T<<<(128 * 128 + T - 1) / T, T, 0, stream>>>(Wf, WT16, 128);
    gat_wmma_gemm<8><<<gemmBlk, T, 0, stream>>>(A16, WT16, XH, rowTiles);
    // attention scalars
    k_att<<<nhBlk, T, 0, stream>>>(XH, as, ad, ASRC, ADST, N);
    // reset reduction buffers
    (void)hipMemsetAsync(MENC, 0, (size_t)N * 4 * 4, stream);
    (void)hipMemsetAsync(DEN,  0, (size_t)N * 4 * 4, stream);
    (void)hipMemsetAsync(ACC,  0, (size_t)N * 128 * 4, stream);
    // segment softmax + weighted scatter
    k_edge_logit<<<ehBlk, T, 0, stream>>>(ei, Er, Et, ASRC, ADST, EXL, MENC);
    k_edge_exp<<<ehBlk, T, 0, stream>>>(ei, Er, Et, EXL, MENC, DEN);
    k_edge_scatter<<<Et, 128, 0, stream>>>(ei, Er, XH, EXL, DEN, ACC);
    // bias + relu -> next layer's f16 input
    k_bias_relu_f16<<<nfBlk, T, 0, stream>>>(ACC, bias, X16, N * 128);
  };

  // layer 1 (input x -> f16)
  k_f32_to_f16<<<nfBlk, T, 0, stream>>>(x, X16, N * 128);
  gat_layer(X16, W1, as1, ad1, b1);
  // layer 2 (X16 now holds relu(h1) in f16)
  gat_layer(X16, W2, as2, ad2, b2);
  // classifier: [N,128] @ [128,48-padded] (WMMA), then log_softmax over 40
  k_pad_wc_T<<<(48 * 128 + T - 1) / T, T, 0, stream>>>(Wc, WCT);
  gat_wmma_gemm<3><<<gemmBlk, T, 0, stream>>>(X16, WCT, LPAD, rowTiles);
  k_logsoftmax<<<(N + T - 1) / T, T, 0, stream>>>(LPAD, bc, out, N);

  (void)n_in; (void)out_size; (void)ws_size;
}
